// Net_10995116278480
// MI455X (gfx1250) — compile-verified, matrix operand order checked by reading, not run on hardware
//
#include <hip/hip_runtime.h>
#include <hip/hip_bf16.h>

typedef __attribute__((ext_vector_type(16))) _Float16 v16h;
typedef __attribute__((ext_vector_type(8)))  _Float16 v8h;
typedef __attribute__((ext_vector_type(8)))  float    v8f;

#define BATCH 8
#define NPTS  2048
#define KNN   64
#define M1    1024
#define M2    256
#define FINF  3.0e38f

// ---------------------------------------------------------------------------
// WMMA fragment helpers (layouts per CDNA5 ISA §7.12.2, wave32)
// ---------------------------------------------------------------------------

// A fragment: 16x32 f16 row-major source, leading dim ld (halfs).
// Per lane the fragment is two contiguous 8-half runs -> two 16B vector loads.
__device__ __forceinline__ v16h load_fragA(const _Float16* A, int ld, int m0, int k0, int lane) {
    int off = (lane >= 16) ? 8 : 0;
    const _Float16* rp = A + (size_t)(m0 + (lane & 15)) * ld + k0 + off;
    v8h lo = *(const v8h*)rp;
    v8h hi = *(const v8h*)(rp + 16);
    return __builtin_shufflevector(lo, hi, 0, 1, 2, 3, 4, 5, 6, 7,
                                   8, 9, 10, 11, 12, 13, 14, 15);
}

// B fragment from weights pre-packed in fragment-major order:
// Wp[((kt*ntn + nt)*32 + lane)*16 + j]  (one aligned 32B load per lane)
__device__ __forceinline__ v16h load_fragB_packed(const _Float16* Wp, int ntn,
                                                  int kt, int nt, int lane) {
    return *(const v16h*)(Wp + (((size_t)kt * ntn + nt) * 32 + lane) * 16);
}

// One fused MLP layer over a 64-row LDS tile: out = relu(in@W + b)*g + be
__device__ void wmma_layer(const _Float16* __restrict__ Wp,
                           const float* __restrict__ bias,
                           const float* __restrict__ gain,
                           const float* __restrict__ beta,
                           const _Float16* inL, _Float16* outL,
                           int CinP, int Cout, int tid) {
    const int wave = tid >> 5, lane = tid & 31;
    const int ntn = Cout >> 4;
    const int ntiles = 4 * ntn;   // 64 rows -> 4 M-tiles
    for (int t = wave; t < ntiles; t += 8) {
        int m0 = (t & 3) * 16;
        int nt = t >> 2;
        v8f acc = {};
        for (int kt = 0; kt < (CinP >> 5); ++kt) {
            v16h a = load_fragA(inL, CinP, m0, kt * 32, lane);
            v16h b = load_fragB_packed(Wp, ntn, kt, nt, lane);
            acc = __builtin_amdgcn_wmma_f32_16x16x32_f16(false, a, false, b,
                                                         (short)0, acc, false, false);
        }
        int N = nt * 16 + (lane & 15);
        int rbase = m0 + ((lane >= 16) ? 8 : 0);
        float bb = bias[N], gg = gain[N], ee = beta[N];
#pragma unroll
        for (int r = 0; r < 8; ++r) {
            float v = fmaxf(acc[r] + bb, 0.0f) * gg + ee;
            outL[(size_t)(rbase + r) * Cout + N] = (_Float16)v;
        }
    }
}

// ---------------------------------------------------------------------------
// Weight repack f32 -> f16, zero-padded K (rowsP), fragment-major layout
// ---------------------------------------------------------------------------
__global__ void pack_w_kernel(const float* __restrict__ W, _Float16* __restrict__ Wp,
                              int rows, int cols, int rowsP) {
    int i = blockIdx.x * 256 + threadIdx.x;
    if (i >= rowsP * cols) return;
    int j    = i & 15;
    int lane = (i >> 4) & 31;
    int tile = i >> 9;
    int ntn = cols >> 4;
    int kt = tile / ntn, nt = tile - kt * ntn;
    int v = j >> 1;
    int k = kt * 32 + (v < 4 ? 2 * v : 16 + 2 * (v - 4)) + ((lane >= 16) ? 8 : 0) + (j & 1);
    int n = nt * 16 + (lane & 15);
    Wp[i] = (k < rows) ? (_Float16)W[k * cols + n] : (_Float16)0.0f;
}

// ---------------------------------------------------------------------------
// Farthest point sampling: one block per cloud (sequential scan, LDS argmax)
// ---------------------------------------------------------------------------
__global__ __launch_bounds__(256) void fps_kernel(const float* __restrict__ pos, int n,
                                                  int stride, int m, int* __restrict__ idx) {
    int b = blockIdx.x;
    const float* P = pos + (size_t)b * n * stride;
    __shared__ float mind[NPTS];
    __shared__ float rv[256];
    __shared__ int   ri[256];
    __shared__ int   s_last;
    int tid = threadIdx.x;
    int npt = n / 256;
    for (int i = 0; i < npt; ++i) mind[tid * npt + i] = 1e10f;
    if (tid == 0) s_last = 0;
    __syncthreads();
    for (int it = 0; it < m; ++it) {
        int last = s_last;
        if (tid == 0) idx[b * m + it] = last;
        float lx = P[last * stride + 0], ly = P[last * stride + 1], lz = P[last * stride + 2];
        float bestv = -1.0f;
        int besti = tid * npt;
        for (int i = 0; i < npt; ++i) {
            int p = tid * npt + i;
            float dx = P[p * stride + 0] - lx;
            float dy = P[p * stride + 1] - ly;
            float dz = P[p * stride + 2] - lz;
            float d = dx * dx + dy * dy + dz * dz;
            float mv = fminf(mind[p], d);
            mind[p] = mv;
            if (mv > bestv) { bestv = mv; besti = p; }
        }
        rv[tid] = bestv; ri[tid] = besti;
        __syncthreads();
        for (int s = 128; s > 0; s >>= 1) {
            if (tid < s) {
                if (rv[tid + s] > rv[tid] ||
                    (rv[tid + s] == rv[tid] && ri[tid + s] < ri[tid])) {
                    rv[tid] = rv[tid + s]; ri[tid] = ri[tid + s];
                }
            }
            __syncthreads();
        }
        if (tid == 0) s_last = ri[0];
        __syncthreads();
    }
}

// ---------------------------------------------------------------------------
// Ball query + top-K nearest: one block per centroid (iterative LDS argmin)
// ---------------------------------------------------------------------------
__global__ __launch_bounds__(256) void group_kernel(const float* __restrict__ pos, int stride,
                                                    int n, const int* __restrict__ idx, int m,
                                                    float r2, int* __restrict__ nidx,
                                                    float* __restrict__ cpos) {
    int c = blockIdx.x;
    int b = c / m;
    const float* P = pos + (size_t)b * n * stride;
    __shared__ float d[NPTS];
    __shared__ float rv[256];
    __shared__ int   ri[256];
    __shared__ float cp[3];
    int tid = threadIdx.x;
    if (tid == 0) {
        int pi = idx[c];
        cp[0] = P[pi * stride + 0];
        cp[1] = P[pi * stride + 1];
        cp[2] = P[pi * stride + 2];
        cpos[c * 3 + 0] = cp[0];
        cpos[c * 3 + 1] = cp[1];
        cpos[c * 3 + 2] = cp[2];
    }
    __syncthreads();
    int npt = n / 256;
    for (int i = 0; i < npt; ++i) {
        int p = tid * npt + i;
        float dx = P[p * stride + 0] - cp[0];
        float dy = P[p * stride + 1] - cp[1];
        float dz = P[p * stride + 2] - cp[2];
        float dd = dx * dx + dy * dy + dz * dz;
        d[p] = (dd <= r2) ? dd : FINF;
    }
    __syncthreads();
    for (int k = 0; k < KNN; ++k) {
        float bestv = 4.0e38f;
        int besti = tid * npt;
        for (int i = 0; i < npt; ++i) {
            int p = tid * npt + i;
            if (d[p] < bestv) { bestv = d[p]; besti = p; }
        }
        rv[tid] = bestv; ri[tid] = besti;
        __syncthreads();
        for (int s = 128; s > 0; s >>= 1) {
            if (tid < s) {
                if (rv[tid + s] < rv[tid] ||
                    (rv[tid + s] == rv[tid] && ri[tid + s] < ri[tid])) {
                    rv[tid] = rv[tid + s]; ri[tid] = ri[tid + s];
                }
            }
            __syncthreads();
        }
        if (tid == 0) {
            if (rv[0] >= 2.9e38f) {
                nidx[c * KNN + k] = -1;
            } else {
                nidx[c * KNN + k] = ri[0];
                d[ri[0]] = FINF;
            }
        }
        __syncthreads();
    }
}

// ---------------------------------------------------------------------------
// Fused SA1: gather (feat3 | rel3, pad 32) -> 32->64->64->128 MLP -> masked max
// ---------------------------------------------------------------------------
__global__ __launch_bounds__(256) void sa1_kernel(
        const float* __restrict__ x, const int* __restrict__ nidx,
        const float* __restrict__ cpos,
        const _Float16* W0, const float* b0, const float* g0, const float* e0,
        const _Float16* W1, const float* b1, const float* g1, const float* e1,
        const _Float16* W2, const float* b2, const float* g2, const float* e2,
        _Float16* __restrict__ h1) {
    int c = blockIdx.x;
    int b = c / M1;
    __shared__ __align__(32) _Float16 sA[64 * 64];    // in0 (stride 32) then t2 (stride 64)
    __shared__ __align__(32) _Float16 sB[64 * 64];    // t1
    __shared__ __align__(32) _Float16 sC[64 * 128];   // out
    __shared__ float cp[3];
    __shared__ int vmask[KNN];
    int tid = threadIdx.x;
    if (tid == 0) { cp[0] = cpos[c * 3]; cp[1] = cpos[c * 3 + 1]; cp[2] = cpos[c * 3 + 2]; }
    __syncthreads();
    if (tid < KNN) {
        int nj = nidx[c * KNN + tid];
        vmask[tid] = (nj >= 0);
        const float* px = x + ((size_t)b * NPTS + (nj >= 0 ? nj : 0)) * 6;
        _Float16* row = sA + tid * 32;
        row[0] = (_Float16)px[3];
        row[1] = (_Float16)px[4];
        row[2] = (_Float16)px[5];
        row[3] = (_Float16)(px[0] - cp[0]);
        row[4] = (_Float16)(px[1] - cp[1]);
        row[5] = (_Float16)(px[2] - cp[2]);
        for (int j = 6; j < 32; ++j) row[j] = (_Float16)0.0f;
    }
    __syncthreads();
    wmma_layer(W0, b0, g0, e0, sA, sB, 32, 64, tid);
    __syncthreads();
    wmma_layer(W1, b1, g1, e1, sB, sA, 64, 64, tid);
    __syncthreads();
    wmma_layer(W2, b2, g2, e2, sA, sC, 64, 128, tid);
    __syncthreads();
    if (tid < 128) {
        float mx = -FINF;
        for (int j = 0; j < KNN; ++j)
            if (vmask[j]) mx = fmaxf(mx, (float)sC[j * 128 + tid]);
        h1[(size_t)c * 128 + tid] = (_Float16)mx;
    }
}

// ---------------------------------------------------------------------------
// Fused SA2: gather via GLOBAL_LOAD_ASYNC_TO_LDS (ASYNCcnt) -> 160->128->128->256
//            -> masked max.  Invalid rows load a clamped valid row; the
//            max-pool mask discards them (only pad cols must be NaN-free).
// ---------------------------------------------------------------------------
__global__ __launch_bounds__(256) void sa2_kernel(
        const _Float16* __restrict__ h1, const float* __restrict__ pos1,
        const int* __restrict__ nidx, const float* __restrict__ cpos,
        const _Float16* W0, const float* b0, const float* g0, const float* e0,
        const _Float16* W1, const float* b1, const float* g1, const float* e1,
        const _Float16* W2, const float* b2, const float* g2, const float* e2,
        _Float16* __restrict__ h2) {
    int c = blockIdx.x;
    int b = c / M2;
    __shared__ __align__(32) _Float16 sA[64 * 160];   // in0 (stride 160), later t2 (stride 128)
    __shared__ __align__(32) _Float16 sB[64 * 256];   // t1 (stride 128), later out (stride 256)
    __shared__ float cp[3];
    __shared__ int vmask[KNN];
    int tid = threadIdx.x;
    if (tid == 0) { cp[0] = cpos[c * 3]; cp[1] = cpos[c * 3 + 1]; cp[2] = cpos[c * 3 + 2]; }
    __syncthreads();
    if (tid < KNN) {
        int nj = nidx[c * KNN + tid];
        vmask[tid] = (nj >= 0);
        int gp = b * M1 + (nj >= 0 ? nj : 0);
        const _Float16* hrow = h1 + (size_t)gp * 128;
        _Float16* row = sA + tid * 160;
        // async DMA 128 f16 (256B) into LDS: 16 x b128, no VGPR round-trip
        unsigned ldsoff = (unsigned)(uintptr_t)row;
#pragma unroll
        for (int j = 0; j < 16; ++j) {
            unsigned dst = ldsoff + j * 16;
            unsigned long long src = (unsigned long long)(uintptr_t)(hrow + j * 8);
            asm volatile("global_load_async_to_lds_b128 %0, %1, off"
                         :: "v"(dst), "v"(src) : "memory");
        }
        const float* pp = pos1 + (size_t)gp * 3;
        row[128] = (_Float16)(pp[0] - cp[0]);
        row[129] = (_Float16)(pp[1] - cp[1]);
        row[130] = (_Float16)(pp[2] - cp[2]);
        for (int j = 131; j < 160; ++j) row[j] = (_Float16)0.0f;
    }
    asm volatile("s_wait_asynccnt 0x0" ::: "memory");
    __syncthreads();
    wmma_layer(W0, b0, g0, e0, sA, sB, 160, 128, tid);
    __syncthreads();
    wmma_layer(W1, b1, g1, e1, sB, sA, 128, 128, tid);
    __syncthreads();
    wmma_layer(W2, b2, g2, e2, sA, sB, 128, 256, tid);
    __syncthreads();
    {
        int col = tid;
        float mx = -FINF;
        for (int j = 0; j < KNN; ++j)
            if (vmask[j]) mx = fmaxf(mx, (float)sB[j * 256 + col]);
        h2[(size_t)c * 256 + col] = (_Float16)mx;
    }
}

// ---------------------------------------------------------------------------
// Generic WMMA GEMM: Out = epi(A[M,CinP] @ W[CinP,Cout] + b)
// mode 0: relu(x+b)*g+be ; mode 1: relu(x+b)
// ---------------------------------------------------------------------------
__global__ __launch_bounds__(256) void gemm_wmma_kernel(
        const _Float16* __restrict__ A, const _Float16* __restrict__ Wp,
        const float* __restrict__ bias, const float* __restrict__ gain,
        const float* __restrict__ beta, _Float16* __restrict__ Out,
        int M, int CinP, int Cout, int mode) {
    int tid = threadIdx.x, lane = tid & 31;
    int wid = blockIdx.x * 8 + (tid >> 5);
    int nw = gridDim.x * 8;
    int mt_n = M >> 4, ntn = Cout >> 4;
    int tiles = mt_n * ntn;
    for (int t = wid; t < tiles; t += nw) {
        int m0 = (t % mt_n) * 16;
        int nt = t / mt_n;
        v8f acc = {};
        for (int kt = 0; kt < (CinP >> 5); ++kt) {
            v16h a = load_fragA(A, CinP, m0, kt * 32, lane);
            v16h b = load_fragB_packed(Wp, ntn, kt, nt, lane);
            acc = __builtin_amdgcn_wmma_f32_16x16x32_f16(false, a, false, b,
                                                         (short)0, acc, false, false);
        }
        int N = nt * 16 + (lane & 15);
        int rbase = m0 + ((lane >= 16) ? 8 : 0);
        float bb = bias[N];
        float gg = (mode == 0) ? gain[N] : 1.0f;
        float ee = (mode == 0) ? beta[N] : 0.0f;
#pragma unroll
        for (int r = 0; r < 8; ++r) {
            float v = fmaxf(acc[r] + bb, 0.0f) * gg + ee;
            Out[(size_t)(rbase + r) * Cout + N] = (_Float16)v;
        }
    }
}

// ---------------------------------------------------------------------------
// MLP3 input build: X3[r, 0:256]=h2[r], [256:259]=cpos2[r], [259:288]=0
// ---------------------------------------------------------------------------
__global__ void x3_build_kernel(const _Float16* __restrict__ h2,
                                const float* __restrict__ cpos2,
                                _Float16* __restrict__ X3) {
    int i = blockIdx.x * 256 + threadIdx.x;
    if (i >= BATCH * M2 * 288) return;
    int r = i / 288, c = i % 288;
    _Float16 v;
    if (c < 256)      v = h2[(size_t)r * 256 + c];
    else if (c < 259) v = (_Float16)cpos2[r * 3 + (c - 256)];
    else              v = (_Float16)0.0f;
    X3[i] = v;
}

// Global max over M2 points per batch; branch 0 writes z, branch 1 adds
__global__ void pointmax_kernel(const _Float16* __restrict__ Y, float* __restrict__ z,
                                int branch) {
    int i = blockIdx.x * 256 + threadIdx.x;
    if (i >= BATCH * 1024) return;
    int b = i >> 10, col = i & 1023;
    const _Float16* base = Y + (size_t)b * M2 * 1024 + col;
    float mx = -FINF;
    for (int p = 0; p < M2; ++p) mx = fmaxf(mx, (float)base[(size_t)p * 1024]);
    if (branch == 0) z[i] = mx; else z[i] += mx;
}

// z (f32, 8x1024) -> zh (f16, 16x1024, rows 8..15 zero)
__global__ void zh_build_kernel(const float* __restrict__ z, _Float16* __restrict__ zh) {
    int i = blockIdx.x * 256 + threadIdx.x;
    if (i >= 16 * 1024) return;
    int r = i >> 10;
    zh[i] = (r < BATCH) ? (_Float16)z[i] : (_Float16)0.0f;
}

// lin3 + log_softmax (tiny): H2 f16 [16x256], W3 f32 [256x5], b3 f32[5]
__global__ void head_kernel(const _Float16* __restrict__ H2, const float* __restrict__ W3,
                            const float* __restrict__ b3, float* __restrict__ out) {
    __shared__ float lg[BATCH][5];
    int tid = threadIdx.x;
    if (tid < BATCH * 5) {
        int r = tid / 5, cc = tid % 5;
        float s = b3[cc];
        for (int k = 0; k < 256; ++k) s += (float)H2[r * 256 + k] * W3[k * 5 + cc];
        lg[r][cc] = s;
    }
    __syncthreads();
    if (tid < BATCH) {
        float mx = lg[tid][0];
        for (int c = 1; c < 5; ++c) mx = fmaxf(mx, lg[tid][c]);
        float se = 0.0f;
        for (int c = 0; c < 5; ++c) se += expf(lg[tid][c] - mx);
        float lse = mx + logf(se);
        for (int c = 0; c < 5; ++c) out[tid * 5 + c] = lg[tid][c] - lse;
    }
}

// ---------------------------------------------------------------------------
// Host launcher
// ---------------------------------------------------------------------------
extern "C" void kernel_launch(void* const* d_in, const int* in_sizes, int n_in,
                              void* d_out, int out_size, void* d_ws, size_t ws_size,
                              hipStream_t stream) {
    (void)in_sizes; (void)n_in; (void)out_size; (void)ws_size;

    char* base = (char*)d_ws;
    size_t off = 0;
    auto alloc = [&](size_t bytes) -> void* {
        void* p = base + off;
        off = (off + bytes + 255) & ~(size_t)255;
        return p;
    };

    // f16 fragment-packed weights (K padded to mult of 32)
    _Float16* Wh1_0 = (_Float16*)alloc((size_t)32 * 64 * 2);
    _Float16* Wh1_1 = (_Float16*)alloc((size_t)64 * 64 * 2);
    _Float16* Wh1_2 = (_Float16*)alloc((size_t)64 * 128 * 2);
    _Float16* Wh2_0 = (_Float16*)alloc((size_t)160 * 128 * 2);
    _Float16* Wh2_1 = (_Float16*)alloc((size_t)128 * 128 * 2);
    _Float16* Wh2_2 = (_Float16*)alloc((size_t)128 * 256 * 2);
    _Float16* Wh3_0 = (_Float16*)alloc((size_t)288 * 256 * 2);
    _Float16* Wh3_1 = (_Float16*)alloc((size_t)256 * 512 * 2);
    _Float16* Wh3_2 = (_Float16*)alloc((size_t)512 * 1024 * 2);
    _Float16* WhL1  = (_Float16*)alloc((size_t)1024 * 512 * 2);
    _Float16* WhL2  = (_Float16*)alloc((size_t)512 * 256 * 2);
    // per-branch intermediates
    int*   idx1  = (int*)alloc((size_t)BATCH * M1 * 4);
    int*   nidx1 = (int*)alloc((size_t)BATCH * M1 * KNN * 4);
    float* cpos1 = (float*)alloc((size_t)BATCH * M1 * 3 * 4);
    int*   idx2  = (int*)alloc((size_t)BATCH * M2 * 4);
    int*   nidx2 = (int*)alloc((size_t)BATCH * M2 * KNN * 4);
    float* cpos2 = (float*)alloc((size_t)BATCH * M2 * 3 * 4);
    _Float16* h1 = (_Float16*)alloc((size_t)BATCH * M1 * 128 * 2);
    _Float16* h2 = (_Float16*)alloc((size_t)BATCH * M2 * 256 * 2);
    _Float16* X3  = (_Float16*)alloc((size_t)BATCH * M2 * 288 * 2);
    _Float16* Y3a = (_Float16*)alloc((size_t)BATCH * M2 * 256 * 2);
    _Float16* Y3b = (_Float16*)alloc((size_t)BATCH * M2 * 512 * 2);
    _Float16* Y3c = (_Float16*)alloc((size_t)BATCH * M2 * 1024 * 2);
    float* z  = (float*)alloc((size_t)BATCH * 1024 * 4);
    _Float16* zh = (_Float16*)alloc((size_t)16 * 1024 * 2);
    _Float16* H1 = (_Float16*)alloc((size_t)16 * 512 * 2);
    _Float16* H2 = (_Float16*)alloc((size_t)16 * 256 * 2);

    // param pointers (setup_inputs dict insertion order)
    const float* W1_[3] = {(const float*)d_in[2],  (const float*)d_in[6],  (const float*)d_in[10]};
    const float* b1_[3] = {(const float*)d_in[3],  (const float*)d_in[7],  (const float*)d_in[11]};
    const float* g1_[3] = {(const float*)d_in[4],  (const float*)d_in[8],  (const float*)d_in[12]};
    const float* e1_[3] = {(const float*)d_in[5],  (const float*)d_in[9],  (const float*)d_in[13]};
    const float* W2_[3] = {(const float*)d_in[14], (const float*)d_in[18], (const float*)d_in[22]};
    const float* b2_[3] = {(const float*)d_in[15], (const float*)d_in[19], (const float*)d_in[23]};
    const float* g2_[3] = {(const float*)d_in[16], (const float*)d_in[20], (const float*)d_in[24]};
    const float* e2_[3] = {(const float*)d_in[17], (const float*)d_in[21], (const float*)d_in[25]};
    const float* W3_[3] = {(const float*)d_in[26], (const float*)d_in[30], (const float*)d_in[34]};
    const float* b3_[3] = {(const float*)d_in[27], (const float*)d_in[31], (const float*)d_in[35]};
    const float* g3_[3] = {(const float*)d_in[28], (const float*)d_in[32], (const float*)d_in[36]};
    const float* e3_[3] = {(const float*)d_in[29], (const float*)d_in[33], (const float*)d_in[37]};
    const float* WL1 = (const float*)d_in[38]; const float* bL1 = (const float*)d_in[39];
    const float* WL2 = (const float*)d_in[40]; const float* bL2 = (const float*)d_in[41];
    const float* WL3 = (const float*)d_in[42]; const float* bL3 = (const float*)d_in[43];

    auto pack = [&](const float* W, _Float16* Wp, int rows, int cols, int rowsP) {
        int n = rowsP * cols;
        pack_w_kernel<<<(n + 255) / 256, 256, 0, stream>>>(W, Wp, rows, cols, rowsP);
    };
    pack(W1_[0], Wh1_0, 6, 64, 32);
    pack(W1_[1], Wh1_1, 64, 64, 64);
    pack(W1_[2], Wh1_2, 64, 128, 64);
    pack(W2_[0], Wh2_0, 131, 128, 160);
    pack(W2_[1], Wh2_1, 128, 128, 128);
    pack(W2_[2], Wh2_2, 128, 256, 128);
    pack(W3_[0], Wh3_0, 259, 256, 288);
    pack(W3_[1], Wh3_1, 256, 512, 256);
    pack(W3_[2], Wh3_2, 512, 1024, 512);
    pack(WL1, WhL1, 1024, 512, 1024);
    pack(WL2, WhL2, 512, 256, 512);

    for (int br = 0; br < 2; ++br) {
        const float* X = (const float*)d_in[br];  // [B, 2048, 6]

        // SA1
        fps_kernel<<<BATCH, 256, 0, stream>>>(X, NPTS, 6, M1, idx1);
        group_kernel<<<BATCH * M1, 256, 0, stream>>>(X, 6, NPTS, idx1, M1,
                                                     0.2f * 0.2f, nidx1, cpos1);
        sa1_kernel<<<BATCH * M1, 256, 0, stream>>>(
            X, nidx1, cpos1,
            Wh1_0, b1_[0], g1_[0], e1_[0],
            Wh1_1, b1_[1], g1_[1], e1_[1],
            Wh1_2, b1_[2], g1_[2], e1_[2], h1);

        // SA2
        fps_kernel<<<BATCH, 256, 0, stream>>>(cpos1, M1, 3, M2, idx2);
        group_kernel<<<BATCH * M2, 256, 0, stream>>>(cpos1, 3, M1, idx2, M2,
                                                     0.4f * 0.4f, nidx2, cpos2);
        sa2_kernel<<<BATCH * M2, 256, 0, stream>>>(
            h1, cpos1, nidx2, cpos2,
            Wh2_0, b2_[0], g2_[0], e2_[0],
            Wh2_1, b2_[1], g2_[1], e2_[1],
            Wh2_2, b2_[2], g2_[2], e2_[2], h2);

        // MLP3 over all B*M2 points
        {
            int n = BATCH * M2 * 288;
            x3_build_kernel<<<(n + 255) / 256, 256, 0, stream>>>(h2, cpos2, X3);
        }
        int Mg = BATCH * M2;  // 2048
        {
            int tiles = (Mg >> 4) * (256 >> 4);
            gemm_wmma_kernel<<<(tiles + 7) / 8, 256, 0, stream>>>(
                X3, Wh3_0, b3_[0], g3_[0], e3_[0], Y3a, Mg, 288, 256, 0);
        }
        {
            int tiles = (Mg >> 4) * (512 >> 4);
            gemm_wmma_kernel<<<(tiles + 7) / 8, 256, 0, stream>>>(
                Y3a, Wh3_1, b3_[1], g3_[1], e3_[1], Y3b, Mg, 256, 512, 0);
        }
        {
            int tiles = (Mg >> 4) * (1024 >> 4);
            gemm_wmma_kernel<<<(tiles + 7) / 8, 256, 0, stream>>>(
                Y3b, Wh3_2, b3_[2], g3_[2], e3_[2], Y3c, Mg, 512, 1024, 0);
        }
        pointmax_kernel<<<(BATCH * 1024 + 255) / 256, 256, 0, stream>>>(Y3c, z, br);
    }

    // Head: lin1 -> lin2 (WMMA, M padded to 16), lin3 + log_softmax
    zh_build_kernel<<<(16 * 1024 + 255) / 256, 256, 0, stream>>>(z, zh);
    {
        int tiles = (16 >> 4) * (512 >> 4);
        gemm_wmma_kernel<<<(tiles + 7) / 8, 256, 0, stream>>>(
            zh, WhL1, bL1, nullptr, nullptr, H1, 16, 1024, 512, 1);
    }
    {
        int tiles = (16 >> 4) * (256 >> 4);
        gemm_wmma_kernel<<<(tiles + 7) / 8, 256, 0, stream>>>(
            H1, WhL2, bL2, nullptr, nullptr, H2, 16, 512, 256, 1);
    }
    head_kernel<<<1, 64, 0, stream>>>(H2, WL3, bL3, (float*)d_out);
}
